// LLaDAAttention_58025008169318
// MI455X (gfx1250) — matrix operand order repaired and checked by reading, not compiled, not run on hardware
//
#include <hip/hip_runtime.h>
#include <cstdint>

// ---- problem constants ----
#define T_CT   2048
#define S_CT   512
#define HID_CT 4096
#define H_CT   32
#define KVH_CT 8
#define D_CT   128
#define G_CT   4
#define HALF_D 64
#define NKV_CT 2048          // 2*KVH*D
#define SCALE_F 0.08838834764831845f   // 1/sqrt(128)
#define THETA_F 10000.0f

typedef __bf16 bf16;
typedef __attribute__((ext_vector_type(16))) __bf16 v16bf;
typedef __attribute__((ext_vector_type(8)))  float  v8f;
typedef __attribute__((ext_vector_type(4)))  unsigned int u32x4;

union FragBF { v16bf v; u32x4 q[2]; };

__device__ __forceinline__ bf16 f2bf(float f) {
  union { float f; unsigned u; } x; x.f = f;
  unsigned r = x.u + 0x7FFFu + ((x.u >> 16) & 1u);
  unsigned short h = (unsigned short)(r >> 16);
  return __builtin_bit_cast(bf16, h);
}

__device__ __forceinline__ v8f wmma_bf16(FragBF a, FragBF b, v8f c) {
  return __builtin_amdgcn_wmma_f32_16x16x32_bf16(false, a.v, false, b.v,
                                                 (short)0, c, false, false);
}

// ---------------- elementwise / layout kernels ----------------

__global__ __launch_bounds__(256)
void convert_f32_bf16_kernel(const float* __restrict__ src, bf16* __restrict__ dst, size_t n) {
  size_t tid = (size_t)blockIdx.x * blockDim.x + threadIdx.x;
  if (tid < n) dst[tid] = f2bf(src[tid]);
}

// W: (K,N) f32 row-major  ->  Wt: (N,K) bf16 row-major
__global__ __launch_bounds__(256)
void transpose_bf16_kernel(const float* __restrict__ W, bf16* __restrict__ Wt, int K, int N) {
  size_t tid = (size_t)blockIdx.x * blockDim.x + threadIdx.x;
  if (tid >= (size_t)K * N) return;
  int k = (int)(tid % K);
  int n = (int)(tid / K);
  Wt[tid] = f2bf(W[(size_t)k * N + n]);
}

// prev_v (T,KVH,D) f32 -> vt (KVH,D,T) bf16
__global__ __launch_bounds__(256)
void fill_vt_kernel(const float* __restrict__ pv, bf16* __restrict__ vt) {
  size_t tid = (size_t)blockIdx.x * blockDim.x + threadIdx.x;
  if (tid >= (size_t)T_CT * KVH_CT * D_CT) return;
  int d = (int)(tid % D_CT);
  size_t th = tid / D_CT;
  int h = (int)(th % KVH_CT);
  int t = (int)(th / KVH_CT);
  vt[((size_t)h * D_CT + d) * T_CT + t] = f2bf(pv[tid]);
}

// q_f32 (T,H,D) -> rope -> q_bf (T,H,D)
__global__ __launch_bounds__(256)
void rope_q_kernel(const float* __restrict__ qf, const int* __restrict__ pos,
                   bf16* __restrict__ qb) {
  size_t tid = (size_t)blockIdx.x * blockDim.x + threadIdx.x;
  if (tid >= (size_t)T_CT * H_CT * HALF_D) return;
  int i = (int)(tid % HALF_D);
  size_t th = tid / HALF_D;
  int h = (int)(th % H_CT);
  int t = (int)(th / H_CT);
  float p  = (float)pos[t];
  float fr = __powf(THETA_F, -(float)i / (float)HALF_D);
  float sn, cs; __sincosf(p * fr, &sn, &cs);
  const float* b = qf + ((size_t)t * H_CT + h) * D_CT;
  float x1 = b[i], x2 = b[i + HALF_D];
  bf16* o = qb + ((size_t)t * H_CT + h) * D_CT;
  o[i]          = f2bf(x1 * cs - x2 * sn);
  o[i + HALF_D] = f2bf(x2 * cs + x1 * sn);
}

// kv_f32 (S, 2*KVH*D): rope K rows and scatter into k_full (T,KVH,D),
// scatter V rows into vt (KVH,D,T)
__global__ __launch_bounds__(256)
void rope_scatter_kv_kernel(const float* __restrict__ kvf, const int* __restrict__ pos,
                            const int* __restrict__ idx,
                            bf16* __restrict__ kfull, bf16* __restrict__ vt) {
  size_t tid = (size_t)blockIdx.x * blockDim.x + threadIdx.x;
  if (tid >= (size_t)S_CT * KVH_CT * HALF_D) return;
  int i = (int)(tid % HALF_D);
  size_t sh = tid / HALF_D;
  int h = (int)(sh % KVH_CT);
  int s = (int)(sh / KVH_CT);
  int t = idx[s];
  float p  = (float)pos[t];
  float fr = __powf(THETA_F, -(float)i / (float)HALF_D);
  float sn, cs; __sincosf(p * fr, &sn, &cs);
  const float* kb = kvf + (size_t)s * NKV_CT + (size_t)h * D_CT;
  float x1 = kb[i], x2 = kb[i + HALF_D];
  bf16* ko = kfull + ((size_t)t * KVH_CT + h) * D_CT;
  ko[i]          = f2bf(x1 * cs - x2 * sn);
  ko[i + HALF_D] = f2bf(x2 * cs + x1 * sn);
  const float* vb = kvf + (size_t)s * NKV_CT + (size_t)(KVH_CT * D_CT) + (size_t)h * D_CT;
  vt[((size_t)h * D_CT + i) * T_CT + t]          = f2bf(vb[i]);
  vt[((size_t)h * D_CT + i + HALF_D) * T_CT + t] = f2bf(vb[i + HALF_D]);
}

// ---------------- WMMA GEMM:  C(M,N) f32 = A(M,K) bf16 @ Bt(N,K)^T ----------------
// block: 256 threads = 8 waves laid out 2(M) x 4(N); block tile 64 x 256.
// wave tile 32(M) x 64(N): 2 A-frags x 4 B-frags -> 8 WMMAs / K-step
// (~21 flop/byte of fragment traffic; all operands L2-resident).
// Optional rowIdx gathers A rows (for hs[idx_salient]).
__global__ __launch_bounds__(256)
void gemm_bf16_kernel(const bf16* __restrict__ A, const int* __restrict__ rowIdx,
                      const bf16* __restrict__ Bt, float* __restrict__ C,
                      int M, int N, int K) {
  const int lane = threadIdx.x & 31;
  const int wave = threadIdx.x >> 5;
  const int half = lane >> 4;
  const int l15  = lane & 15;
  const int m0 = blockIdx.y * 64 + (wave & 1) * 32;
  const int n0 = blockIdx.x * 256 + (wave >> 1) * 64;

  const int arow0 = m0 + l15;
  const int arow1 = m0 + 16 + l15;
  const int asrc0 = rowIdx ? rowIdx[arow0] : arow0;
  const int asrc1 = rowIdx ? rowIdx[arow1] : arow1;
  const bf16* Ar0 = A + (size_t)asrc0 * K;
  const bf16* Ar1 = A + (size_t)asrc1 * K;
  const bf16* Bp[4];
#pragma unroll
  for (int j = 0; j < 4; ++j) Bp[j] = Bt + (size_t)(n0 + j * 16 + l15) * K;

  v8f c0[4], c1[4];
#pragma unroll
  for (int j = 0; j < 4; ++j) {
    c0[j] = (v8f){0.f,0.f,0.f,0.f,0.f,0.f,0.f,0.f};
    c1[j] = (v8f){0.f,0.f,0.f,0.f,0.f,0.f,0.f,0.f};
  }

  for (int k = 0; k < K; k += 32) {
    FragBF a0, a1, b[4];
    const int ka = k + half * 8;    // A: lanes 0-15 -> K[k..k+8)+K[k+16..k+24); lanes 16-31 -> +8
    a0.q[0] = *(const u32x4*)(Ar0 + ka);
    a0.q[1] = *(const u32x4*)(Ar0 + ka + 16);
    a1.q[0] = *(const u32x4*)(Ar1 + ka);
    a1.q[1] = *(const u32x4*)(Ar1 + ka + 16);
    const int kb = k + half * 8;    // B: lanes 0-15 -> K[k..k+8)+K[k+16..k+24); lanes 16-31 -> +8
#pragma unroll
    for (int j = 0; j < 4; ++j) {
      b[j].q[0] = *(const u32x4*)(Bp[j] + kb);
      b[j].q[1] = *(const u32x4*)(Bp[j] + kb + 16);
    }
#pragma unroll
    for (int j = 0; j < 4; ++j) {
      c0[j] = wmma_bf16(a0, b[j], c0[j]);
      c1[j] = wmma_bf16(a1, b[j], c1[j]);
    }
  }
  // C layout: VGPR r, lane -> (M = tile_m + r + 8*half, N = tile_n + l15)
#pragma unroll
  for (int j = 0; j < 4; ++j) {
    float* Cr0 = C + (size_t)(m0 + half * 8) * N + n0 + j * 16 + l15;
    float* Cr1 = C + (size_t)(m0 + 16 + half * 8) * N + n0 + j * 16 + l15;
#pragma unroll
    for (int r = 0; r < 8; ++r) {
      Cr0[(size_t)r * N] = c0[j][r];
      Cr1[(size_t)r * N] = c1[j][r];
    }
  }
}

// ---------------- GQA flash attention ----------------
// grid: (H, T/16/4), block 128 = 4 waves; one wave = one 16-query tile of one head.
// Qb (T,H,D) bf16, Kf (T,KVH,D) bf16, Vt (KVH,D,T) bf16, Ob (T,H*D) bf16.
__global__ __launch_bounds__(128)
void attn_kernel(const bf16* __restrict__ Qb, const bf16* __restrict__ Kf,
                 const bf16* __restrict__ Vt, bf16* __restrict__ Ob) {
  __shared__ __align__(16) bf16 plds[4][16 * 32];

  const int lane = threadIdx.x & 31;
  const int wave = threadIdx.x >> 5;
  const int half = lane >> 4;
  const int l15  = lane & 15;
  const int head = blockIdx.x;
  const int h_kv = head / G_CT;
  const int qt   = blockIdx.y * 4 + wave;
  const int q0   = qt * 16;

  // Q fragments covering D=128 (4 x K=32)
  FragBF qa[4];
  const bf16* Qrow = Qb + ((size_t)(q0 + l15) * H_CT + head) * D_CT;
#pragma unroll
  for (int kc = 0; kc < 4; ++kc) {
    const int ka = kc * 32 + half * 8;
    qa[kc].q[0] = *(const u32x4*)(Qrow + ka);
    qa[kc].q[1] = *(const u32x4*)(Qrow + ka + 16);
  }

  v8f o[8];
#pragma unroll
  for (int dt = 0; dt < 8; ++dt) o[dt] = (v8f){0.f,0.f,0.f,0.f,0.f,0.f,0.f,0.f};
  float m_run[8], l_run[8];
#pragma unroll
  for (int r = 0; r < 8; ++r) { m_run[r] = -3.4e38f; l_run[r] = 0.f; }

  const size_t kstride = (size_t)KVH_CT * D_CT;
  const bf16* Kbase = Kf + (size_t)h_kv * D_CT;

  for (int k0 = 0; k0 < T_CT; k0 += 32) {
    // ---- scores: Q(16xD) x K^T -> two 16x16 tiles (32 keys) ----
    v8f s0 = {0.f,0.f,0.f,0.f,0.f,0.f,0.f,0.f};
    v8f s1 = {0.f,0.f,0.f,0.f,0.f,0.f,0.f,0.f};
    const bf16* K0 = Kbase + (size_t)(k0 + l15) * kstride;
    const bf16* K1 = Kbase + (size_t)(k0 + 16 + l15) * kstride;
#pragma unroll
    for (int kc = 0; kc < 4; ++kc) {
      FragBF b0, b1;
      const int kb = kc * 32 + half * 8;
      b0.q[0] = *(const u32x4*)(K0 + kb);
      b0.q[1] = *(const u32x4*)(K0 + kb + 16);
      b1.q[0] = *(const u32x4*)(K1 + kb);
      b1.q[1] = *(const u32x4*)(K1 + kb + 16);
      s0 = wmma_bf16(qa[kc], b0, s0);
      s1 = wmma_bf16(qa[kc], b1, s1);
    }
    // ---- online softmax (rows live in 16-lane halves) ----
#pragma unroll
    for (int r = 0; r < 8; ++r) {
      float a0 = s0[r] * SCALE_F;
      float a1 = s1[r] * SCALE_F;
      float tmax = fmaxf(a0, a1);
#pragma unroll
      for (int off = 8; off >= 1; off >>= 1)
        tmax = fmaxf(tmax, __shfl_xor(tmax, off, 32));
      float mnew = fmaxf(m_run[r], tmax);
      float corr = __expf(m_run[r] - mnew);
      m_run[r] = mnew;
      float p0 = __expf(a0 - mnew);
      float p1 = __expf(a1 - mnew);
      float ps = p0 + p1;
#pragma unroll
      for (int off = 8; off >= 1; off >>= 1)
        ps += __shfl_xor(ps, off, 32);
      l_run[r] = l_run[r] * corr + ps;
#pragma unroll
      for (int dt = 0; dt < 8; ++dt) o[dt][r] *= corr;
      // stage P (16x32) row-major in LDS for re-layout to A-fragment
      plds[wave][(r + 8 * half) * 32 + l15]      = f2bf(p0);
      plds[wave][(r + 8 * half) * 32 + l15 + 16] = f2bf(p1);
    }
    __syncthreads();
    FragBF pa;
    {
      const bf16* prow = &plds[wave][l15 * 32 + half * 8];
      pa.q[0] = *(const u32x4*)(prow);
      pa.q[1] = *(const u32x4*)(prow + 16);
    }
    // ---- O += P(16x32) x V(32x128) ----
#pragma unroll
    for (int dt = 0; dt < 8; ++dt) {
      FragBF vb;
      const bf16* vrow = Vt + ((size_t)h_kv * D_CT + dt * 16 + l15) * T_CT + k0 + half * 8;
      vb.q[0] = *(const u32x4*)(vrow);
      vb.q[1] = *(const u32x4*)(vrow + 16);
      o[dt] = wmma_bf16(pa, vb, o[dt]);
    }
    __syncthreads();
  }
  // ---- epilogue: normalize and store bf16 (T, H*D) ----
#pragma unroll
  for (int dt = 0; dt < 8; ++dt) {
#pragma unroll
    for (int r = 0; r < 8; ++r) {
      float val = o[dt][r] / l_run[r];
      Ob[(size_t)(q0 + r + 8 * half) * HID_CT + head * D_CT + dt * 16 + l15] = f2bf(val);
    }
  }
}

// ---------------- host launch ----------------

static inline unsigned blocks_for(size_t n) { return (unsigned)((n + 255) / 256); }

extern "C" void kernel_launch(void* const* d_in, const int* in_sizes, int n_in,
                              void* d_out, int out_size, void* d_ws, size_t ws_size,
                              hipStream_t stream) {
  const float* hs  = (const float*)d_in[0];
  const float* pk  = (const float*)d_in[1];
  const float* pv  = (const float*)d_in[2];
  const float* Wq  = (const float*)d_in[3];
  const float* Wkv = (const float*)d_in[4];
  const float* Wo  = (const float*)d_in[5];
  const int*   pos = (const int*)d_in[6];
  const int*   idx = (const int*)d_in[7];
  float* out = (float*)d_out;
  (void)in_sizes; (void)n_in; (void)out_size; (void)ws_size;

  char* w = (char*)d_ws;
  bf16*  Wq_t   = (bf16*)w;  w += (size_t)HID_CT * HID_CT * 2;
  bf16*  Wkv_t  = (bf16*)w;  w += (size_t)NKV_CT * HID_CT * 2;
  bf16*  Wo_t   = (bf16*)w;  w += (size_t)HID_CT * HID_CT * 2;
  bf16*  hs_b   = (bf16*)w;  w += (size_t)T_CT * HID_CT * 2;
  float* q_f    = (float*)w; w += (size_t)T_CT * HID_CT * 4;
  bf16*  q_b    = (bf16*)w;  w += (size_t)T_CT * HID_CT * 2;
  float* kv_f   = (float*)w; w += (size_t)S_CT * NKV_CT * 4;
  bf16*  k_full = (bf16*)w;  w += (size_t)T_CT * KVH_CT * D_CT * 2;
  bf16*  v_t    = (bf16*)w;  w += (size_t)T_CT * KVH_CT * D_CT * 2;
  bf16*  attn_b = (bf16*)w;  w += (size_t)T_CT * HID_CT * 2;

  // weight transposes f32 -> bf16 (N-major for contiguous B fragments)
  transpose_bf16_kernel<<<blocks_for((size_t)HID_CT * HID_CT), 256, 0, stream>>>(Wq,  Wq_t,  HID_CT, HID_CT);
  transpose_bf16_kernel<<<blocks_for((size_t)HID_CT * NKV_CT), 256, 0, stream>>>(Wkv, Wkv_t, HID_CT, NKV_CT);
  transpose_bf16_kernel<<<blocks_for((size_t)HID_CT * HID_CT), 256, 0, stream>>>(Wo,  Wo_t,  HID_CT, HID_CT);
  // activations / caches to bf16
  convert_f32_bf16_kernel<<<blocks_for((size_t)T_CT * HID_CT), 256, 0, stream>>>(hs, hs_b, (size_t)T_CT * HID_CT);
  convert_f32_bf16_kernel<<<blocks_for((size_t)T_CT * KVH_CT * D_CT), 256, 0, stream>>>(pk, k_full, (size_t)T_CT * KVH_CT * D_CT);
  fill_vt_kernel<<<blocks_for((size_t)T_CT * KVH_CT * D_CT), 256, 0, stream>>>(pv, v_t);

  // q = hs @ Wq   (T x HID) x (HID x HID)
  {
    dim3 g(HID_CT / 256, T_CT / 64);
    gemm_bf16_kernel<<<g, 256, 0, stream>>>(hs_b, nullptr, Wq_t, q_f, T_CT, HID_CT, HID_CT);
  }
  // kv = hs[idx] @ Wkv   (S x HID) x (HID x 2048)
  {
    dim3 g(NKV_CT / 256, S_CT / 64);
    gemm_bf16_kernel<<<g, 256, 0, stream>>>(hs_b, idx, Wkv_t, kv_f, S_CT, NKV_CT, HID_CT);
  }
  // RoPE + scatter
  rope_q_kernel<<<blocks_for((size_t)T_CT * H_CT * HALF_D), 256, 0, stream>>>(q_f, pos, q_b);
  rope_scatter_kv_kernel<<<blocks_for((size_t)S_CT * KVH_CT * HALF_D), 256, 0, stream>>>(
      kv_f, pos, idx, k_full, v_t);

  // attention
  {
    dim3 g(H_CT, T_CT / 16 / 4);
    attn_kernel<<<g, 128, 0, stream>>>(q_b, k_full, v_t, attn_b);
  }
  // out = attn @ Wo   (T x 4096) x (4096 x 4096) -> f32
  {
    dim3 g(HID_CT / 256, T_CT / 64);
    gemm_bf16_kernel<<<g, 256, 0, stream>>>(attn_b, nullptr, Wo_t, out, T_CT, HID_CT, HID_CT);
  }
}
